// ThreeLayerFCModel_68015102099870
// MI455X (gfx1250) — matrix operand order repaired and checked by reading, not compiled, not run on hardware
//
#include <hip/hip_runtime.h>
#include <stdint.h>

// ---------------------------------------------------------------------------
// Problem constants (match reference)
// ---------------------------------------------------------------------------
#define BATCH 4096
#define IN_DIM 1024
#define D_DIM 4096
#define S1_DIM 8192
#define S2_DIM 8192

// GEMM tiling: block tile 256x128, 8 waves (wave32), wave tile 64x64
// = 4x4 WMMA 16x16x32 accumulators per wave.
#define BM 256
#define BN 128
#define BK 32
#define LDSS 40   // LDS row stride in bf16 elements (32 + 8 pad, keeps 16B align)

typedef __attribute__((ext_vector_type(16))) __bf16 v16bf;
typedef __attribute__((ext_vector_type(8)))  float  v8f;
typedef __attribute__((ext_vector_type(4)))  __bf16 v4bf;

union Frag16 {
    v16bf v;
    uint4 q[2];
};

// ---------------------------------------------------------------------------
// gfx1250 async global->LDS copy path (ASYNCcnt-tracked), with sync fallback
// ---------------------------------------------------------------------------
#if defined(__AMDGCN__) && \
    __has_builtin(__builtin_amdgcn_global_load_async_to_lds_b128) && \
    __has_builtin(__builtin_amdgcn_s_wait_asynccnt)
#define USE_ASYNC 1
#else
#define USE_ASYNC 0
#endif

// Builtin prototype (from clang diagnostic): takes pointers to
// 'int __attribute__((vector_size(16)))' in AS(1) (global) and AS(3) (LDS).
typedef int v4i_gcc __attribute__((vector_size(16)));
typedef __attribute__((address_space(1))) v4i_gcc gv4i;
typedef __attribute__((address_space(3))) v4i_gcc lv4i;

__device__ __forceinline__ void async_cp16(const __bf16* g, __bf16* l)
{
#if USE_ASYNC
    // Low 32 bits of a generic LDS pointer are the LDS byte offset.
    __builtin_amdgcn_global_load_async_to_lds_b128(
        (gv4i*)(uintptr_t)g, (lv4i*)(uint32_t)(uintptr_t)l, 0, 0);
#else
    *(uint4*)l = *(const uint4*)g;
#endif
}

template<int N>
__device__ __forceinline__ void wait_async()
{
#if USE_ASYNC
    __builtin_amdgcn_s_wait_asynccnt(N);
#endif
}

// ---------------------------------------------------------------------------
// f32 -> bf16 conversion (grid-stride, 4 elements / thread / step)
// ---------------------------------------------------------------------------
__global__ __launch_bounds__(256) void f32_to_bf16_kernel(
    const float* __restrict__ src, __bf16* __restrict__ dst, long n4)
{
    long i = (long)blockIdx.x * blockDim.x + threadIdx.x;
    long stride = (long)gridDim.x * blockDim.x;
    const float4* s4 = (const float4*)src;
    v4bf* d4 = (v4bf*)dst;
    for (; i < n4; i += stride) {
        float4 v = s4[i];
        v4bf o;
        o[0] = (__bf16)v.x;
        o[1] = (__bf16)v.y;
        o[2] = (__bf16)v.z;
        o[3] = (__bf16)v.w;
        d4[i] = o;
    }
}

// ---------------------------------------------------------------------------
// C[M,N] = ReLU(A[M,K] @ W[N,K]^T (+ bias))      A,W bf16; acc f32
//
// Double-buffered LDS staging via async global->LDS b128 copies.
// Fragment layouts per CDNA5 ISA 7.12.2 (wave32):
//   A (16x32 bf16): lane<16 row=lane%16 holds K0..7 (v0-3), K16..23 (v4-7);
//                   lane>=16 holds K8..15, K24..31.
//   B (32x16 bf16): lane<16 col=lane%16 holds K0..7, K16..23;
//                   lane>=16 holds K8..15, K24..31.
// ---------------------------------------------------------------------------
template<bool HAS_BIAS, bool OUT_BF16>
__global__ __launch_bounds__(256, 1) void gemm_bias_relu_kernel(
    const __bf16* __restrict__ A, const __bf16* __restrict__ W,
    const float* __restrict__ bias, void* __restrict__ out,
    int M, int N, int K)
{
    __shared__ __bf16 As[2][BM * LDSS];   // 2 x 256 x 32 bf16 (+pad)
    __shared__ __bf16 Bs[2][BN * LDSS];   // 2 x 128 x 32 bf16 (+pad)

    const int tid  = threadIdx.x;
    const int lane = tid & 31;
    const int wave = tid >> 5;
    const int waveM = (wave & 3) * 64;   // 4 waves along M
    const int waveN = (wave >> 2) * 64;  // 2 waves along N
    const int blockM = blockIdx.x * BM;
    const int blockN = blockIdx.y * BN;
    const int half = lane >> 4;          // 0: lanes 0-15, 1: lanes 16-31
    const int l15  = lane & 15;

    v8f acc[4][4];
#pragma unroll
    for (int tm = 0; tm < 4; ++tm)
#pragma unroll
        for (int tn = 0; tn < 4; ++tn)
#pragma unroll
            for (int r = 0; r < 8; ++r)
                acc[tm][tn][r] = 0.0f;

    // Issue one BK tile: A = 1024 x 16B chunks, B = 512 x 16B chunks,
    // 256 threads -> 6 async copies per thread.
    auto fill = [&](int bufi, int kk) {
#pragma unroll
        for (int c = 0; c < 4; ++c) {
            int chunk = tid + c * 256;
            int row = chunk >> 2;        // 0..255
            int g   = chunk & 3;         // 0..3 (8 bf16 each)
            async_cp16(A + (size_t)(blockM + row) * K + kk + g * 8,
                       &As[bufi][row * LDSS + g * 8]);
        }
#pragma unroll
        for (int c = 0; c < 2; ++c) {
            int chunk = tid + c * 256;
            int row = chunk >> 2;        // 0..127
            int g   = chunk & 3;
            async_cp16(W + (size_t)(blockN + row) * K + kk + g * 8,
                       &Bs[bufi][row * LDSS + g * 8]);
        }
    };

    fill(0, 0);
    int buf = 0;

    for (int k0 = 0; k0 < K; k0 += BK) {
        const bool more = (k0 + BK) < K;
        if (more) {
            fill(buf ^ 1, k0 + BK);   // prefetch next tile into other buffer
            wait_async<6>();          // first 6 outstanding = current buffer
        } else {
            wait_async<0>();
        }
        __syncthreads();              // all waves' chunks visible in LDS

        const __bf16* Ab = As[buf];
        const __bf16* Bb = Bs[buf];

        v16bf afrag[4], bfrag[4];
#pragma unroll
        for (int tm = 0; tm < 4; ++tm) {
            int row = waveM + tm * 16 + l15;
            Frag16 f;
            f.q[0] = *(const uint4*)(Ab + row * LDSS + half * 8);
            f.q[1] = *(const uint4*)(Ab + row * LDSS + half * 8 + 16);
            afrag[tm] = f.v;
        }
#pragma unroll
        for (int tn = 0; tn < 4; ++tn) {
            int col = waveN + tn * 16 + l15;
            Frag16 f;
            f.q[0] = *(const uint4*)(Bb + col * LDSS + half * 8);
            f.q[1] = *(const uint4*)(Bb + col * LDSS + half * 8 + 16);
            bfrag[tn] = f.v;
        }

#pragma unroll
        for (int tm = 0; tm < 4; ++tm)
#pragma unroll
            for (int tn = 0; tn < 4; ++tn)
                acc[tm][tn] = __builtin_amdgcn_wmma_f32_16x16x32_bf16(
                    false, afrag[tm], false, bfrag[tn],
                    (short)0, acc[tm][tn], false, false);

        buf ^= 1;
        __syncthreads();              // reads of 'buf^1' done before its refill
    }

    // Epilogue: bias + ReLU, store bf16 (intermediate) or f32 (final).
#pragma unroll
    for (int tn = 0; tn < 4; ++tn) {
        int col = blockN + waveN + tn * 16 + l15;
        float bv = HAS_BIAS ? bias[col] : 0.0f;
#pragma unroll
        for (int tm = 0; tm < 4; ++tm) {
            int rbase = blockM + waveM + tm * 16 + half * 8;
#pragma unroll
            for (int r = 0; r < 8; ++r) {
                float v = acc[tm][tn][r] + bv;
                v = v > 0.0f ? v : 0.0f;
                size_t idx = (size_t)(rbase + r) * N + col;
                if (OUT_BF16)
                    ((__bf16*)out)[idx] = (__bf16)v;
                else
                    ((float*)out)[idx] = v;
            }
        }
    }
}

// ---------------------------------------------------------------------------
// Launcher
// ---------------------------------------------------------------------------
extern "C" void kernel_launch(void* const* d_in, const int* in_sizes, int n_in,
                              void* d_out, int out_size, void* d_ws, size_t ws_size,
                              hipStream_t stream)
{
    const float* x  = (const float*)d_in[0];   // [B, IN]
    const float* Wd = (const float*)d_in[1];   // [D, IN]
    const float* bd = (const float*)d_in[2];   // [D]
    const float* W1 = (const float*)d_in[3];   // [S1, D]
    const float* W2 = (const float*)d_in[4];   // [S2, S1]
    float* out = (float*)d_out;                // [B, S2]

    // Workspace layout (all bf16, 256B aligned by construction)
    char* ws = (char*)d_ws;
    __bf16* xb  = (__bf16*)ws;  ws += (size_t)BATCH * IN_DIM * 2;   // 8 MB
    __bf16* Wdb = (__bf16*)ws;  ws += (size_t)D_DIM * IN_DIM * 2;   // 8 MB
    __bf16* W1b = (__bf16*)ws;  ws += (size_t)S1_DIM * D_DIM * 2;   // 64 MB
    __bf16* W2b = (__bf16*)ws;  ws += (size_t)S2_DIM * S1_DIM * 2;  // 128 MB
    __bf16* x1b = (__bf16*)ws;  ws += (size_t)BATCH * D_DIM * 2;    // 32 MB
    __bf16* x2b = (__bf16*)ws;  ws += (size_t)BATCH * S1_DIM * 2;   // 64 MB

    // 1) One-shot f32 -> bf16 conversion of inputs + weights
    auto conv = [&](const float* s, __bf16* d, long n) {
        long n4 = n / 4;
        int blocks = (int)((n4 + 255) / 256);
        if (blocks > 65536) blocks = 65536;
        f32_to_bf16_kernel<<<blocks, 256, 0, stream>>>(s, d, n4);
    };
    conv(x,  xb,  (long)BATCH * IN_DIM);
    conv(Wd, Wdb, (long)D_DIM * IN_DIM);
    conv(W1, W1b, (long)S1_DIM * D_DIM);
    conv(W2, W2b, (long)S2_DIM * S1_DIM);

    // 2) Layer 1: x1 = relu(x @ Wd^T + bd)   [4096, 4096], bf16 out
    gemm_bias_relu_kernel<true, true><<<dim3(BATCH / BM, D_DIM / BN), 256, 0, stream>>>(
        xb, Wdb, bd, x1b, BATCH, D_DIM, IN_DIM);

    // 3) Layer 2: x2 = relu(x1 @ W1^T)       [4096, 8192], bf16 out
    gemm_bias_relu_kernel<false, true><<<dim3(BATCH / BM, S1_DIM / BN), 256, 0, stream>>>(
        x1b, W1b, nullptr, x2b, BATCH, S1_DIM, D_DIM);

    // 4) Layer 3: out = relu(x2 @ W2^T)      [4096, 8192], f32 out
    gemm_bias_relu_kernel<false, false><<<dim3(BATCH / BM, S2_DIM / BN), 256, 0, stream>>>(
        x2b, W2b, nullptr, out, BATCH, S2_DIM, S1_DIM);
}
